// SentenceT5_MLP_Agg_60438779789383
// MI455X (gfx1250) — matrix operand order repaired and checked by reading, not compiled, not run on hardware
//
#include <hip/hip_runtime.h>
#include <hip/hip_bf16.h>
#include <math.h>

// ---------------------------------------------------------------------------
// SentenceT5 MLP + noisy-OR aggregation for MI455X (gfx1250, wave32, WMMA,
// async global->LDS copies).
//
// Roofline: GEMM1 (Qproj hoisted) 38.7 GFLOP + GEMM2 19.3 GFLOP, ~150 MB of
// HBM traffic -> heavily compute bound -> bf16 WMMA (K=32/issue) with fp32
// accumulate. Intermediates (H1 48MB, H2 24MB bf16) are L2-resident (192MB).
//
// Pipeline:
//   K0: Qproj[b,n] = questions[b] @ W1[0:768] + b1          (tiny fp32 VALU)
//   K1: segments fp32 -> bf16 (one pass, enables pure-copy async staging)
//   K2: W1b / W2 fp32 -> pre-swizzled padded bf16 B-tile images
//   K3: H1 = gelu(Sbf @ W1b + Qproj[bag])   async-LDS + double-buffered WMMA
//   K4: H2 = gelu(H1 @ W2 + b2)             async-LDS + double-buffered WMMA
//   K5: preds = softmax(H2 @ W3 + b3)       wave-per-row, shfl reduce
//   K6: bag aggregation (prefix/suffix cumprod leave-one-out)
// ---------------------------------------------------------------------------

#define BB 128
#define SS 256
#define EE 768
#define CC 4
#define MM (BB * SS)

// Flip to 0 if the async inline-asm mnemonics need fixing.
#define USE_ASYNC_LDS 1

typedef __attribute__((ext_vector_type(16))) __bf16 v16bf;
typedef __attribute__((ext_vector_type(8)))  float  v8f;

__device__ __forceinline__ unsigned bfpack2(float a, float b) {
  unsigned ua = __float_as_uint(a);
  unsigned ub = __float_as_uint(b);
  ua = (ua + 0x7FFFu + ((ua >> 16) & 1u)) >> 16;   // RNE fp32 -> bf16
  ub = (ub + 0x7FFFu + ((ub >> 16) & 1u)) >> 16;
  return ua | (ub << 16);
}
__device__ __forceinline__ unsigned short f2bf(float a) {
  unsigned ua = __float_as_uint(a);
  return (unsigned short)((ua + 0x7FFFu + ((ua >> 16) & 1u)) >> 16);
}
__device__ __forceinline__ float bf2f(unsigned short h) {
  return __uint_as_float(((unsigned)h) << 16);
}
__device__ __forceinline__ float gelu_exact(float v) {
  return 0.5f * v * (1.0f + erff(v * 0.70710678118654752f));
}

// Flat pointers to LDS carry the LDS byte offset in their low 32 bits
// (aperture rule).
__device__ __forceinline__ unsigned lds_byte_addr(const void* p) {
  return (unsigned)(size_t)p;
}

#if USE_ASYNC_LDS
// GLOBAL_LOAD_ASYNC_TO_LDS_B128: per-lane 16B copy, tracked by ASYNCcnt.
__device__ __forceinline__ void async_b128(unsigned lds_byte, const void* gsrc) {
  asm volatile("global_load_async_to_lds_b128 %0, %1, off"
               :
               : "v"(lds_byte), "v"(gsrc)
               : "memory");
}
__device__ __forceinline__ void wait_async0() {
  asm volatile("s_wait_asynccnt 0x0" ::: "memory");
}
#else
__device__ __forceinline__ void async_b128(unsigned lds_byte, const void* gsrc) {
  // Fallback: plain global_load + ds_store through VGPRs.
  *(uint4*)((size_t)lds_byte | ((size_t)(&lds_byte) & 0UL)) = *(const uint4*)gsrc; // unused
}
__device__ __forceinline__ void wait_async0() {}
#endif

// ---------------------------------------------------------------------------
// K0: Qproj = questions @ W1[0:768,:] + b1   (128x768 output, trivial cost)
// ---------------------------------------------------------------------------
__global__ __launch_bounds__(256) void qproj_kernel(
    const float* __restrict__ q, const float* __restrict__ W1,
    const float* __restrict__ b1, float* __restrict__ Qp) {
  int b = blockIdx.x;
  __shared__ float qs[EE];
  for (int k = threadIdx.x; k < EE; k += 256) qs[k] = q[(size_t)b * EE + k];
  __syncthreads();
  for (int n = threadIdx.x; n < EE; n += 256) {
    float acc = b1[n];
    for (int k = 0; k < EE; ++k)
      acc = fmaf(qs[k], W1[(size_t)k * EE + n], acc);
    Qp[(size_t)b * EE + n] = acc;
  }
}

// ---------------------------------------------------------------------------
// K1: fp32 -> bf16 row-major conversion (one packed pair per thread)
// ---------------------------------------------------------------------------
__global__ __launch_bounds__(256) void f32_to_bf16_kernel(
    const float* __restrict__ src, unsigned* __restrict__ dst, int npairs) {
  int id = blockIdx.x * 256 + threadIdx.x;
  if (id >= npairs) return;
  float2 f = ((const float2*)src)[id];
  dst[id] = bfpack2(f.x, f.y);
}

// ---------------------------------------------------------------------------
// K2: pre-swizzle W [Ktotal, Ntotal] fp32 into contiguous padded LDS B-tile
// images: tile (kt, nt) holds bytes [n*144 + kp*4] = packed bf16 pair
// (W[k0+2kp, n0+n], W[k0+2kp+1, n0+n]), kp in 0..31 (K-tile 64), n in 0..127.
// 144B row pad keeps the GEMM's ds_load_b128 fragment loads conflict-free.
// ---------------------------------------------------------------------------
#define BTILE_BYTES 18432   // 128 * 144
__global__ __launch_bounds__(256) void wswizzle_kernel(
    const float* __restrict__ W, unsigned char* __restrict__ Wp,
    int Ntotal, int Ktotal) {
  int id = blockIdx.x * 256 + threadIdx.x;
  int total = (Ktotal >> 1) * Ntotal;
  if (id >= total) return;
  int kpg = id / Ntotal;         // global pair row (k/2)
  int ng  = id % Ntotal;
  int kt  = kpg >> 5;            // 32 pairs per K-tile of 64
  int kp  = kpg & 31;
  int nt  = ng >> 7;
  int n   = ng & 127;
  int k   = kpg * 2;
  unsigned pk = bfpack2(W[(size_t)k * Ntotal + ng],
                        W[(size_t)(k + 1) * Ntotal + ng]);
  int NT = Ntotal >> 7;
  *(unsigned*)(Wp + (size_t)(kt * NT + nt) * BTILE_BYTES + n * 144 + kp * 4) = pk;
}

// ---------------------------------------------------------------------------
// K3/K4: tiled GEMM + bias + exact-GELU, bf16 WMMA fp32-acc.
//   block tile 128x128, K-tile 64 (two WMMA K-steps), double-buffered LDS,
//   8 waves; wave tile 32(M)x64(N) -> 16 v_wmma per K-tile per wave.
//   All staging is async global->LDS 16B copies (no VGPR round-trip).
// ---------------------------------------------------------------------------
template <bool QPROJ_BIAS>
__global__ __launch_bounds__(256) void mlp_gemm_gelu(
    const unsigned short* __restrict__ Abf,   // [M, Ktotal] bf16 row-major
    const unsigned char* __restrict__ Wp,     // pre-swizzled B tiles
    const float* __restrict__ bias,           // QPROJ: [B, Ntotal]; else [Ntotal]
    unsigned short* __restrict__ Out,         // [M, Ntotal] bf16
    int Ntotal, int Ktotal) {
  constexpr int AROW = 144;
  constexpr int TILE = 128 * AROW;            // 18432 B
  __shared__ __align__(16) unsigned char smem[4 * TILE];  // A0 A1 B0 B1 = 72KB

  const int tid  = threadIdx.x;
  const int bn   = blockIdx.x;
  const int bm   = blockIdx.y;
  const int wave = tid >> 5;
  const int lane = tid & 31;
  const int wm   = (wave & 3) * 32;
  const int wn   = (wave >> 2) * 64;
  const int hl   = lane & 15;
  const int khi  = lane >> 4;
  const int NT   = Ntotal >> 7;
  const int nkt  = Ktotal >> 6;

  const unsigned ldsA0 = lds_byte_addr(smem);
  const unsigned ldsB0 = ldsA0 + 2u * TILE;

  v8f acc[2][4] = {};
  union Frag { uint4 u[2]; v16bf v; };

  auto stage = [&](int kt, int buf) {
    const unsigned la = ldsA0 + buf * TILE;
    const unsigned lb = ldsB0 + buf * TILE;
    // B tile: one contiguous pre-swizzled 18432B image -> 1152 16B chunks
    const unsigned char* gB = Wp + (size_t)(kt * NT + bn) * BTILE_BYTES;
#if USE_ASYNC_LDS
#pragma unroll
    for (int i = 0; i < 4; ++i) {
      int id = tid + i * 256;
      async_b128(lb + id * 16, gB + (size_t)id * 16);
    }
    if (tid < 128) {
      int id = 1024 + tid;
      async_b128(lb + id * 16, gB + (size_t)id * 16);
    }
    // A tile: 128 rows x 64 bf16 (8 chunks of 16B per row), padded rows
    const unsigned char* gA =
        (const unsigned char*)(Abf + (size_t)(bm * 128) * Ktotal + kt * 64);
#pragma unroll
    for (int i = 0; i < 4; ++i) {
      int id = tid + i * 256;
      int r = id >> 3, c = id & 7;
      async_b128(la + r * AROW + c * 16,
                 gA + (size_t)r * Ktotal * 2 + c * 16);
    }
#else
    unsigned char* sB = smem + 2 * TILE + buf * TILE;
#pragma unroll
    for (int i = 0; i < 4; ++i) {
      int id = tid + i * 256;
      *(uint4*)(sB + id * 16) = *(const uint4*)(gB + (size_t)id * 16);
    }
    if (tid < 128) {
      int id = 1024 + tid;
      *(uint4*)(sB + id * 16) = *(const uint4*)(gB + (size_t)id * 16);
    }
    unsigned char* sA = smem + buf * TILE;
    const unsigned char* gA =
        (const unsigned char*)(Abf + (size_t)(bm * 128) * Ktotal + kt * 64);
#pragma unroll
    for (int i = 0; i < 4; ++i) {
      int id = tid + i * 256;
      int r = id >> 3, c = id & 7;
      *(uint4*)(sA + r * AROW + c * 16) =
          *(const uint4*)(gA + (size_t)r * Ktotal * 2 + c * 16);
    }
#endif
  };

  stage(0, 0);
  wait_async0();
  __syncthreads();

  for (int kt = 0; kt < nkt; ++kt) {
    const int cur = kt & 1;
    if (kt + 1 < nkt) stage(kt + 1, cur ^ 1);   // fill other buffer async

    const unsigned char* sA = smem + cur * TILE;
    const unsigned char* sB = smem + (2 + cur) * TILE;
#pragma unroll
    for (int s = 0; s < 2; ++s) {               // two K=32 steps per tile
      Frag afr[2], bfr[4];
#pragma unroll
      for (int mf = 0; mf < 2; ++mf) {
        const unsigned char* p =
            sA + (wm + mf * 16 + hl) * AROW + s * 64 + khi * 16;
        afr[mf].u[0] = *(const uint4*)p;
        afr[mf].u[1] = *(const uint4*)(p + 32);
      }
#pragma unroll
      for (int nf = 0; nf < 4; ++nf) {
        const unsigned char* p =
            sB + (wn + nf * 16 + hl) * AROW + s * 64 + khi * 16;
        bfr[nf].u[0] = *(const uint4*)p;
        bfr[nf].u[1] = *(const uint4*)(p + 32);
      }
#pragma unroll
      for (int mf = 0; mf < 2; ++mf)
#pragma unroll
        for (int nf = 0; nf < 4; ++nf)
          acc[mf][nf] = __builtin_amdgcn_wmma_f32_16x16x32_bf16(
              false, afr[mf].v, false, bfr[nf].v, (short)0, acc[mf][nf],
              false, false);
    }
    if (kt + 1 < nkt) wait_async0();
    __syncthreads();
  }

  // ---- epilogue: bias + exact GELU -> bf16 store -------------------------
#pragma unroll
  for (int mf = 0; mf < 2; ++mf) {
#pragma unroll
    for (int nf = 0; nf < 4; ++nf) {
      int gnn = bn * 128 + wn + nf * 16 + hl;
#pragma unroll
      for (int r = 0; r < 8; ++r) {
        int gm = bm * 128 + wm + mf * 16 + r + khi * 8;  // lanes16-31: M += 8
        float v = acc[mf][nf][r];
        float bv;
        if constexpr (QPROJ_BIAS)
          bv = bias[(size_t)(gm >> 8) * Ntotal + gnn];   // bag = row / 256
        else
          bv = bias[gnn];
        v = gelu_exact(v + bv);
        Out[(size_t)gm * Ntotal + gnn] = f2bf(v);
      }
    }
  }
}

// ---------------------------------------------------------------------------
// K5: preds = softmax(H2 @ W3 + b3); one wave per row (wave32 shfl reduce)
// ---------------------------------------------------------------------------
__global__ __launch_bounds__(256) void head_softmax_kernel(
    const unsigned short* __restrict__ H2, const float* __restrict__ W3,
    const float* __restrict__ b3, float* __restrict__ preds) {
  int row  = blockIdx.x * 8 + (threadIdx.x >> 5);
  int lane = threadIdx.x & 31;
  float a0 = 0.f, a1 = 0.f, a2 = 0.f, a3 = 0.f;
  const unsigned short* h = H2 + (size_t)row * (EE / 2);
  for (int k = lane; k < EE / 2; k += 32) {
    float hv = bf2f(h[k]);
    const float4 w = *(const float4*)(W3 + k * 4);
    a0 = fmaf(hv, w.x, a0);
    a1 = fmaf(hv, w.y, a1);
    a2 = fmaf(hv, w.z, a2);
    a3 = fmaf(hv, w.w, a3);
  }
#pragma unroll
  for (int off = 16; off > 0; off >>= 1) {
    a0 += __shfl_xor(a0, off, 32);
    a1 += __shfl_xor(a1, off, 32);
    a2 += __shfl_xor(a2, off, 32);
    a3 += __shfl_xor(a3, off, 32);
  }
  if (lane == 0) {
    float z0 = a0 + b3[0], z1 = a1 + b3[1], z2 = a2 + b3[2], z3 = a3 + b3[3];
    float m  = fmaxf(fmaxf(z0, z1), fmaxf(z2, z3));
    float e0 = expf(z0 - m), e1 = expf(z1 - m), e2 = expf(z2 - m),
          e3 = expf(z3 - m);
    float inv = 1.0f / (e0 + e1 + e2 + e3);
    float4 r;
    r.x = e0 * inv; r.y = e1 * inv; r.z = e2 * inv; r.w = e3 * inv;
    *(float4*)(preds + (size_t)row * 4) = r;
  }
}

// ---------------------------------------------------------------------------
// K6: bag aggregation — exclusive class-cumsum, leave-one-out segment
// products via prefix/suffix cumprods. One thread per (bag, class).
// ---------------------------------------------------------------------------
__global__ void aggregate_kernel(const float* __restrict__ preds,
                                 const int* __restrict__ num_segments,
                                 float* __restrict__ suf_ws,
                                 float* __restrict__ out) {
  int tid = blockIdx.x * blockDim.x + threadIdx.x;
  if (tid >= BB * CC) return;
  int b = tid >> 2, c = tid & 3;
  int ns = num_segments[b];
  const float* P = preds + (size_t)b * SS * CC;
  float* suf = suf_ws + (size_t)tid * SS;

  float run = 1.0f;
  for (int i = SS - 1; i >= 0; --i) {
    suf[i] = run;
    float sm = 1.0f;
    if (i < ns) {
      const float4 p = *(const float4*)(P + i * 4);
      float s = 0.0f;
      if (c > 0) s += p.x;
      if (c > 1) s += p.y;
      if (c > 2) s += p.z;
      sm = s;
    }
    run *= sm;
  }
  float pre = 1.0f, cpl = 1.0f, sum_t = 0.0f, prod_t = 1.0f;
  for (int i = 0; i < SS; ++i) {
    const float4 p = *(const float4*)(P + i * 4);
    float pc = (c == 0) ? p.x : (c == 1) ? p.y : (c == 2) ? p.z : p.w;
    float sm = 1.0f;
    if (i < ns) {
      float s = 0.0f;
      if (c > 0) s += p.x;
      if (c > 1) s += p.y;
      if (c > 2) s += p.z;
      sm = s;
    }
    cpl *= pre * suf[i];
    if (i < ns) {
      sum_t = fmaf(pc, cpl, sum_t);
      prod_t *= pc;
    }
    pre *= sm;
  }
  out[tid] = sum_t * (1.0f / CC) + prod_t;
}

// ---------------------------------------------------------------------------
extern "C" void kernel_launch(void* const* d_in, const int* in_sizes, int n_in,
                              void* d_out, int out_size, void* d_ws,
                              size_t ws_size, hipStream_t stream) {
  const float* questions    = (const float*)d_in[0];
  const float* segments     = (const float*)d_in[1];
  const float* W1           = (const float*)d_in[2];
  const float* b1           = (const float*)d_in[3];
  const float* W2           = (const float*)d_in[4];
  const float* b2           = (const float*)d_in[5];
  const float* W3           = (const float*)d_in[6];
  const float* b3           = (const float*)d_in[7];
  const int*   num_segments = (const int*)d_in[8];
  float* out = (float*)d_out;

  char* ws = (char*)d_ws;
  size_t off = 0;
  auto alloc = [&](size_t bytes) -> void* {
    void* p = ws + off;
    off = (off + bytes + 255) & ~(size_t)255;
    return p;
  };
  float*          Qp    = (float*)alloc((size_t)BB * EE * 4);            // 0.4 MB
  unsigned*       Sbf   = (unsigned*)alloc((size_t)MM * EE * 2);         // 48 MB
  unsigned short* H1    = (unsigned short*)alloc((size_t)MM * EE * 2);   // 48 MB
  unsigned short* H2    = (unsigned short*)alloc((size_t)MM * (EE/2)*2); // 24 MB
  unsigned char*  W1p   = (unsigned char*)alloc((size_t)(EE/64)*(EE/128)*BTILE_BYTES);
  unsigned char*  W2p   = (unsigned char*)alloc((size_t)(EE/64)*((EE/2)/128)*BTILE_BYTES);
  float*          preds = (float*)alloc((size_t)MM * CC * 4);            // 0.5 MB
  float*          suf   = (float*)alloc((size_t)BB * CC * SS * 4);       // 0.5 MB

  // K0: question projection (per-bag bias for GEMM1)
  qproj_kernel<<<BB, 256, 0, stream>>>(questions, W1, b1, Qp);

  // K1: segments fp32 -> bf16 (pure-copy async staging downstream)
  {
    int npairs = MM * EE / 2;                       // 12.58M
    f32_to_bf16_kernel<<<(npairs + 255) / 256, 256, 0, stream>>>(
        segments, Sbf, npairs);
  }
  // K2: weight pre-swizzle into padded B-tile images
  {
    int t1 = (EE / 2) * EE;                         // W1b pairs
    wswizzle_kernel<<<(t1 + 255) / 256, 256, 0, stream>>>(
        W1 + (size_t)EE * EE, W1p, EE, EE);
    int t2 = (EE / 2) * (EE / 2);                   // W2 pairs
    wswizzle_kernel<<<(t2 + 255) / 256, 256, 0, stream>>>(
        W2, W2p, EE / 2, EE);
  }

  // K3: H1 = gelu(Sbf @ W1b + Qproj[bag])
  mlp_gemm_gelu<true><<<dim3(EE / 128, MM / 128), 256, 0, stream>>>(
      (const unsigned short*)Sbf, W1p, Qp, H1, EE, EE);

  // K4: H2 = gelu(H1 @ W2 + b2)
  mlp_gemm_gelu<false><<<dim3((EE / 2) / 128, MM / 128), 256, 0, stream>>>(
      H1, W2p, b2, H2, EE / 2, EE);

  // K5: preds = softmax(H2 @ W3 + b3)
  head_softmax_kernel<<<MM / 8, 256, 0, stream>>>(H2, W3, b3, preds);

  // K6: per-bag noisy-OR aggregation
  aggregate_kernel<<<2, 256, 0, stream>>>(preds, num_segments, suf, out);
}
